// MAFlow_9165460209844
// MI455X (gfx1250) — compile-verified, matrix operand order checked by reading, not run on hardware
//
#include <hip/hip_runtime.h>

typedef _Float16 h16 __attribute__((ext_vector_type(16)));
typedef _Float16 h8  __attribute__((ext_vector_type(8)));
typedef float    f8  __attribute__((ext_vector_type(8)));

#define DFEAT 45
#define HDIM  256
#define NBLK  6
#define LHID  2
#define KBINS 8
#define MPAR  23      // 3*K-1
#define NOUT  1035    // D*M
#define NPADO 1088    // 17*64, padded output GEMM width
#define KA0   64      // padded input K (45 z + 1 cond + 18 zero)
#define BSTR  40      // LDS B-stage row stride in halves (32 data + 8 pad, bank spread)

// ---------------- weight prep: apply MADE masks, convert to f16 ----------------
__global__ void prep_weights(const float* __restrict__ W0, const float* __restrict__ Wc,
                             const float* __restrict__ Wh, const float* __restrict__ Wf,
                             _Float16* __restrict__ w0c, _Float16* __restrict__ wh,
                             _Float16* __restrict__ wf)
{
    int i = blockIdx.x * 256 + threadIdx.x;
    const int T0 = NBLK * HDIM * KA0;
    const int T1 = NBLK * LHID * HDIM * HDIM;
    const int T2 = NBLK * NPADO * HDIM;
    if (i < T0) {
        int b = i / (HDIM * KA0), r = (i / KA0) % HDIM, c = i % KA0;
        float v = 0.f;
        if (c < DFEAT)       v = ((r % 44) >= c) ? W0[(b * HDIM + r) * DFEAT + c] : 0.f;
        else if (c == DFEAT) v = Wc[b * HDIM + r];
        w0c[i] = (_Float16)v;
    } else if (i < T0 + T1) {
        int j = i - T0;
        int r = (j / HDIM) % HDIM, c = j % HDIM;
        wh[j] = (_Float16)(((r % 44) >= (c % 44)) ? Wh[j] : 0.f);
    } else if (i < T0 + T1 + T2) {
        int j = i - T0 - T1;
        int b = j / (NPADO * HDIM), o = (j / HDIM) % NPADO, c = j % HDIM;
        float v = 0.f;
        if (o < NOUT) v = ((o / MPAR) > (c % 44)) ? Wf[((size_t)b * NOUT + o) * HDIM + c] : 0.f;
        wf[j] = (_Float16)v;
    }
}

// ---------------- init: z = x, build padded f16 A-input, clear logdet ----------------
__global__ void init_state(const float* __restrict__ x, const float* __restrict__ cond,
                           int Brows, _Float16* __restrict__ zA,
                           float* __restrict__ z, float* __restrict__ ld)
{
    int i = blockIdx.x * 256 + threadIdx.x;
    if (i >= Brows * KA0) return;
    int row = i / KA0, c = i % KA0;
    float v = 0.f;
    if (c < DFEAT)       { v = x[row * DFEAT + c]; z[row * DFEAT + c] = v; }
    else if (c == DFEAT) { v = cond[row]; }
    zA[i] = (_Float16)v;
    if (c == 0) ld[row] = 0.f;
}

__device__ __forceinline__ h16 pack16(h8 lo, h8 hi) {
    h16 r;
    #pragma unroll
    for (int q = 0; q < 8; q++) { r[q] = lo[q]; r[q + 8] = hi[q]; }
    return r;
}

// ---- WMMA GEMM: C[M,N] = A[M,K] * Bw^T, Bw stored [Npad][K].
// Workgroup: 8 waves = 256M x 64N; wave = 32M x 64N; B staged via double-buffered LDS.
template<bool RELU, bool OUTF32>
__global__ __launch_bounds__(256) void gemm_wmma(
    const _Float16* __restrict__ A, int lda,
    const _Float16* __restrict__ Bw,
    const float* __restrict__ bias1,
    const float* __restrict__ bias2,
    void* __restrict__ Cout, int ldc,
    int Mrows, int Ncols, int Npad, int K)
{
    __shared__ _Float16 bstage[2][64 * BSTR];

    const int t = threadIdx.x;
    const int lane = t & 31;
    const int wave = t >> 5;
    const int tiles_n = Npad >> 6;
    const int mbase = (blockIdx.x / tiles_n) << 8;   // 256 rows per workgroup
    const int nbase = (blockIdx.x % tiles_n) << 6;

    // cooperative B-stage fill: thread t -> (n = t/4, k8 = (t%4)*8), one b128 each
    const int fn = t >> 2;
    const int fk = (t & 3) << 3;
    const _Float16* bsrc = Bw + (size_t)(nbase + fn) * K + fk;
    _Float16* bdst[2] = { &bstage[0][fn * BSTR + fk], &bstage[1][fn * BSTR + fk] };

    // A fragment pointers: two 16-row subtiles per wave
    // layout (16-bit A 16x32): lane m = lane&15, k = (i&7) + 8*(lane>>4) + 16*(i>>3)
    const int koff = (lane >> 4) << 3;
    const int mrow = mbase + wave * 32;
    const _Float16* aptr0 = A + (size_t)(mrow + (lane & 15)) * lda + koff;
    const _Float16* aptr1 = aptr0 + (size_t)16 * lda;

    // B fragment LDS read offset: lane n = lane&15 within a 16-col subtile
    const int brd = (lane & 15) * BSTR + koff;

    f8 acc[2][4] = {};

    const int S = K >> 5;
    // prologue: fill stage 0
    { h8 v = *(const h8*)bsrc; *(h8*)bdst[0] = v; }
    __syncthreads();

    for (int s = 0; s < S; s++) {
        const int kb = s << 5;
        const bool hasNext = (s + 1) < S;
        h8 pre;
        if (hasNext) pre = *(const h8*)(bsrc + kb + 32);

        // A fragments (direct, contiguous b128 pairs)
        h16 a0 = pack16(*(const h8*)(aptr0 + kb), *(const h8*)(aptr0 + kb + 16));
        h16 a1 = pack16(*(const h8*)(aptr1 + kb), *(const h8*)(aptr1 + kb + 16));

        // B fragments from LDS, 2-slot register rotation to keep ds_loads ahead
        const _Float16* bufp = &bstage[s & 1][0];
        h8 c0lo = *(const h8*)(bufp + brd);
        h8 c0hi = *(const h8*)(bufp + brd + 16);
        #pragma unroll
        for (int sb = 0; sb < 4; sb++) {
            h8 c1lo, c1hi;
            if (sb < 3) {
                const _Float16* bp = bufp + (sb + 1) * (16 * BSTR) + brd;
                c1lo = *(const h8*)bp;
                c1hi = *(const h8*)(bp + 16);
            }
            h16 bb = pack16(c0lo, c0hi);
            acc[0][sb] = __builtin_amdgcn_wmma_f32_16x16x32_f16(
                false, a0, false, bb, (short)0, acc[0][sb], false, false);
            acc[1][sb] = __builtin_amdgcn_wmma_f32_16x16x32_f16(
                false, a1, false, bb, (short)0, acc[1][sb], false, false);
            if (sb < 3) { c0lo = c1lo; c0hi = c1hi; }
        }

        if (hasNext) *(h8*)bdst[(s + 1) & 1] = pre;
        __syncthreads();
    }

    // epilogue; C/D layout: lane n = lane&15, row m = r + 8*(lane>>4)
    const int nlo = lane & 15, mhalf = (lane >> 4) << 3;
    #pragma unroll
    for (int mi = 0; mi < 2; mi++) {
        #pragma unroll
        for (int sb = 0; sb < 4; sb++) {
            int n = nbase + sb * 16 + nlo;
            float bsum = 0.f;
            if (n < Ncols) { bsum = bias1[n]; if (bias2) bsum += bias2[n]; }
            #pragma unroll
            for (int r = 0; r < 8; r++) {
                int m = mrow + mi * 16 + mhalf + r;
                float v = acc[mi][sb][r] + bsum;
                if (RELU) v = v > 0.f ? v : 0.f;
                if (OUTF32) ((float*)Cout)[(size_t)m * ldc + n] = v;
                else        ((_Float16*)Cout)[(size_t)m * ldc + n] = (_Float16)v;
            }
        }
    }
}

// ---------------- rational-quadratic spline + log-det accumulation ----------------
__global__ void spline_kernel(const float* __restrict__ params, int ldp,
                              float* __restrict__ z, _Float16* __restrict__ zA,
                              float* __restrict__ ld, int Brows)
{
    __shared__ float srow[4];
    int t = threadIdx.x;
    int d = t & 63, rl = t >> 6;
    int row = blockIdx.x * 4 + rl;
    if (d == 0) srow[rl] = 0.f;
    __syncthreads();
    if (row < Brows && d < DFEAT) {
        const float TAILF = 13.815510557964274f;     // -log(1e-6)
        const float MINW  = 1e-3f;
        const float PADC  = 0.5397656559f;           // log(exp(1-1e-3)-1)
        const float ISQH  = 0.0625f;                 // 1/sqrt(256)
        const float* p = params + (size_t)row * ldp + d * MPAR;
        float x = z[(size_t)row * DFEAT + d];
        bool inside = (x >= -TAILF) && (x <= TAILF);
        float xc = fminf(fmaxf(x, -TAILF), TAILF);

        float cwv[9], wv[8], chv[9], hv[8], dv[9];
        {
            float u[8], mx = -1e30f;
            #pragma unroll
            for (int i = 0; i < 8; i++) { u[i] = p[i] * ISQH; mx = fmaxf(mx, u[i]); }
            float e[8], s = 0.f;
            #pragma unroll
            for (int i = 0; i < 8; i++) { e[i] = __expf(u[i] - mx); s += e[i]; }
            float sc = (1.f - 8.f * MINW) / s, acc = 0.f;
            cwv[0] = -TAILF;
            #pragma unroll
            for (int i = 0; i < 8; i++) { acc += MINW + sc * e[i]; cwv[i + 1] = 2.f * TAILF * acc - TAILF; }
            cwv[8] = TAILF;
            #pragma unroll
            for (int i = 0; i < 8; i++) wv[i] = cwv[i + 1] - cwv[i];
        }
        {
            float u[8], mx = -1e30f;
            #pragma unroll
            for (int i = 0; i < 8; i++) { u[i] = p[8 + i] * ISQH; mx = fmaxf(mx, u[i]); }
            float e[8], s = 0.f;
            #pragma unroll
            for (int i = 0; i < 8; i++) { e[i] = __expf(u[i] - mx); s += e[i]; }
            float sc = (1.f - 8.f * MINW) / s, acc = 0.f;
            chv[0] = -TAILF;
            #pragma unroll
            for (int i = 0; i < 8; i++) { acc += MINW + sc * e[i]; chv[i + 1] = 2.f * TAILF * acc - TAILF; }
            chv[8] = TAILF;
            #pragma unroll
            for (int i = 0; i < 8; i++) hv[i] = chv[i + 1] - chv[i];
        }
        dv[0] = MINW + logf(1.f + __expf(PADC));
        dv[8] = dv[0];
        #pragma unroll
        for (int j = 1; j < 8; j++) dv[j] = MINW + logf(1.f + __expf(p[16 + j - 1]));

        int idx = 0;
        #pragma unroll
        for (int i = 1; i < 9; i++) idx += (xc >= cwv[i]) ? 1 : 0;
        idx = idx > 7 ? 7 : idx;

        float icw = cwv[idx], ibw = wv[idx], ich = chv[idx], ihh = hv[idx];
        float idelta = hv[idx] / wv[idx];
        float id0 = dv[idx], id1 = dv[idx + 1];
        float th  = (xc - icw) / ibw;
        float th1 = th * (1.f - th);
        float den = idelta + (id0 + id1 - 2.f * idelta) * th1;
        float y   = ich + ihh * (idelta * th * th + id0 * th1) / den;
        float omt = 1.f - th;
        float num = idelta * idelta * (id1 * th * th + 2.f * idelta * th1 + id0 * omt * omt);
        float lad = __logf(num) - 2.f * __logf(den);

        float yo = inside ? y : x;
        z [(size_t)row * DFEAT + d] = yo;
        zA[(size_t)row * KA0  + d] = (_Float16)yo;    // cols 45..63 stay from init
        atomicAdd(&srow[rl], inside ? lad : 0.f);
    }
    __syncthreads();
    if (d == 0 && row < Brows) ld[row] += srow[rl];
}

// ---------------- finalize: standard-normal logprob + logdet ----------------
__global__ void finalize_kernel(const float* __restrict__ z, const float* __restrict__ ld,
                                float* __restrict__ out, int Brows)
{
    int row = blockIdx.x * 256 + threadIdx.x;
    if (row >= Brows) return;
    const float* zr = z + (size_t)row * DFEAT;
    float s = 0.f;
    #pragma unroll 5
    for (int d = 0; d < DFEAT; d++) { float v = zr[d]; s += v * v; }
    out[row] = -0.5f * s - 41.352233994210266f + ld[row];  // 0.5*45*log(2*pi)
}

extern "C" void kernel_launch(void* const* d_in, const int* in_sizes, int n_in,
                              void* d_out, int out_size, void* d_ws, size_t ws_size,
                              hipStream_t stream)
{
    const float* x    = (const float*)d_in[0];
    const float* cond = (const float*)d_in[1];
    const float* W0   = (const float*)d_in[2];
    const float* b0   = (const float*)d_in[3];
    const float* Wc   = (const float*)d_in[4];
    const float* bc   = (const float*)d_in[5];
    const float* Wh   = (const float*)d_in[6];
    const float* bh   = (const float*)d_in[7];
    const float* Wf   = (const float*)d_in[8];
    const float* bf   = (const float*)d_in[9];
    const int Brows = in_sizes[0] / DFEAT;

    char* wsb = (char*)d_ws;
    size_t off = 0;
    auto alloc = [&](size_t bytes) -> char* {
        char* p = wsb + off; off += (bytes + 255) & ~(size_t)255; return p;
    };
    _Float16* w0c    = (_Float16*)alloc((size_t)NBLK * HDIM * KA0 * 2);
    _Float16* whm    = (_Float16*)alloc((size_t)NBLK * LHID * HDIM * HDIM * 2);
    _Float16* wfm    = (_Float16*)alloc((size_t)NBLK * NPADO * HDIM * 2);
    _Float16* zA     = (_Float16*)alloc((size_t)Brows * KA0 * 2);
    _Float16* act0   = (_Float16*)alloc((size_t)Brows * HDIM * 2);
    _Float16* act1   = (_Float16*)alloc((size_t)Brows * HDIM * 2);
    float*    paramsB= (float*)   alloc((size_t)Brows * NPADO * 4);
    float*    zbuf   = (float*)   alloc((size_t)Brows * DFEAT * 4);
    float*    ldbuf  = (float*)   alloc((size_t)Brows * 4);

    {
        int total = NBLK * HDIM * KA0 + NBLK * LHID * HDIM * HDIM + NBLK * NPADO * HDIM;
        prep_weights<<<(total + 255) / 256, 256, 0, stream>>>(W0, Wc, Wh, Wf, w0c, whm, wfm);
    }
    init_state<<<(Brows * KA0 + 255) / 256, 256, 0, stream>>>(x, cond, Brows, zA, zbuf, ldbuf);

    auto gemm = [&](const _Float16* A, int lda, const _Float16* Bm,
                    const float* bi1, const float* bi2, void* C, int ldc,
                    int Ncols, int Npad, int K, bool relu, bool outf32) {
        int blocks = (Brows / 256) * (Npad / 64);   // Brows divisible by 256
        if (outf32)
            gemm_wmma<false, true ><<<blocks, 256, 0, stream>>>(A, lda, Bm, bi1, bi2, C, ldc, Brows, Ncols, Npad, K);
        else if (relu)
            gemm_wmma<true,  false><<<blocks, 256, 0, stream>>>(A, lda, Bm, bi1, bi2, C, ldc, Brows, Ncols, Npad, K);
        else
            gemm_wmma<false, false><<<blocks, 256, 0, stream>>>(A, lda, Bm, bi1, bi2, C, ldc, Brows, Ncols, Npad, K);
    };

    for (int b = 0; b < NBLK; b++) {
        // input layer: z(+cond) -> hidden  (no activation)
        gemm(zA, KA0, w0c + (size_t)b * HDIM * KA0,
             b0 + b * HDIM, bc + b * HDIM, act0, HDIM, HDIM, HDIM, KA0, false, false);
        // 2 hidden layers: linear -> relu
        gemm(act0, HDIM, whm + (size_t)(b * LHID + 0) * HDIM * HDIM,
             bh + (b * LHID + 0) * HDIM, nullptr, act1, HDIM, HDIM, HDIM, HDIM, true, false);
        gemm(act1, HDIM, whm + (size_t)(b * LHID + 1) * HDIM * HDIM,
             bh + (b * LHID + 1) * HDIM, nullptr, act0, HDIM, HDIM, HDIM, HDIM, true, false);
        // output layer: spline params (f32)
        gemm(act0, HDIM, wfm + (size_t)b * NPADO * HDIM,
             bf + (size_t)b * NOUT, nullptr, paramsB, NPADO, NOUT, NPADO, HDIM, false, true);
        // spline transform + logdet
        spline_kernel<<<(Brows + 3) / 4, 256, 0, stream>>>(paramsB, NPADO, zbuf, zA, ldbuf, Brows);
    }
    finalize_kernel<<<(Brows + 255) / 256, 256, 0, stream>>>(zbuf, ldbuf, (float*)d_out, Brows);
}
